// SchemaTemplate_87351044866609
// MI455X (gfx1250) — matrix operand order repaired and checked by reading, not compiled, not run on hardware
//
#include <hip/hip_runtime.h>
#include <hip/hip_bf16.h>
#include <math.h>

// ---------------------------------------------------------------------------
// Shapes: B=16, L=2048, D=512, R=8, H=512.  M = B*L = 32768.
// d_out: bindings (B,R,D)=65536 floats, then fit_score (B,1)=16 floats.
// d_ws : scores_part (4 * B*R*L = 1048576 f32) | role_bias (R*H = 4096 f32)
// ---------------------------------------------------------------------------

typedef __attribute__((ext_vector_type(16))) _Float16 v16h;
typedef __attribute__((ext_vector_type(8)))  _Float16 v8h;
typedef __attribute__((ext_vector_type(4)))  _Float16 v4h;
typedef __attribute__((ext_vector_type(8)))  float    v8f;

union V16H { v16h v; v8h h[2]; };

__device__ __forceinline__ float gelu_exact(float v) {
    return 0.5f * v * (1.0f + erff(v * 0.7071067811865476f));
}

// ---------------------------------------------------------------------------
// Kernel 1: role_bias[r,h] = sum_d role_emb[r,d] * W1[r, D+d, h] + b1[r,h]
// grid (8,2), 256 threads.
// ---------------------------------------------------------------------------
__global__ void k_rolebias(const float* __restrict__ role_emb,
                           const float* __restrict__ W1,
                           const float* __restrict__ b1,
                           float* __restrict__ role_bias) {
    const int r = blockIdx.x;
    const int h = blockIdx.y * 256 + threadIdx.x;
    float acc = b1[r * 512 + h];
    const float* re = role_emb + r * 512;
    const float* w  = W1 + ((size_t)r * 1024 + 512) * 512 + h;
    #pragma unroll 4
    for (int d = 0; d < 512; ++d)
        acc += re[d] * w[(size_t)d * 512];
    role_bias[r * 512 + h] = acc;
}

// ---------------------------------------------------------------------------
// Kernel 2 (dominant): fused  scores[b,r,l] = sum_h gelu(x@W1x + rb)*W2
// WMMA f32_16x16x32_f16, double-buffered LDS pipeline.
// Grid: (M/128, N/128=4, R=8), 256 threads = 8 waves (2 Mgroups x 4 Ngroups);
// each wave: 4 Mtiles x 2 Ntiles. Deterministic: per-N-block partial buffers,
// intra-workgroup combine via LDS in fixed order (no float atomics).
// ---------------------------------------------------------------------------
__global__ void __launch_bounds__(256)
k_score_gemm(const float* __restrict__ x,          // (B*L, 512)
             const float* __restrict__ W1,         // (R, 1024, 512); k < 512
             const float* __restrict__ role_bias,  // (R, 512)
             const float* __restrict__ W2,         // (R, 512)
             float* __restrict__ scores_part) {    // (4, B, R, L)
    __shared__ __align__(16) _Float16 ldsX[2][128 * 32]; // [buf][row][k]
    __shared__ __align__(16) _Float16 ldsW[2][128 * 32]; // [buf][n][k]
    __shared__ float ldsRed[4][128];                     // [waveN][row]

    const int tid   = threadIdx.x;
    const int lane  = tid & 31;
    const int wave  = tid >> 5;
    const int waveM = wave & 1;    // 0..1 -> 64-row group
    const int waveN = wave >> 1;   // 0..3 -> 32-col group
    const int r     = blockIdx.z;
    const int n0    = blockIdx.y * 128;
    const size_t m0 = (size_t)blockIdx.x * 128;
    const float* Wr = W1 + (size_t)r * 1024 * 512 + n0;

    v8f acc[4][2];
    #pragma unroll
    for (int m = 0; m < 4; ++m)
        #pragma unroll
        for (int n = 0; n < 2; ++n)
            acc[m][n] = (v8f){0.f,0.f,0.f,0.f,0.f,0.f,0.f,0.f};

    const int kbA = (lane & 16) ? 8  : 0;  // A halves {kb..kb+7, kb+16..kb+23}
    const int kbB = (lane & 16) ? 8  : 0;  // B halves {kb..kb+7, kb+16..kb+23}

    // staging registers
    float4 xf[4];
    float  wf[4][4];
    // x slots: fi = i*256+tid -> row = fi>>3, c4 = (fi&7)*4
    // W slots: fi = i*256+tid -> nn = fi&127, kk = (fi>>7)*4 (4 floats along k)

    auto load_tiles = [&](int k0) {
        #pragma unroll
        for (int i = 0; i < 4; ++i) {
            int fi  = i * 256 + tid;
            int row = fi >> 3;
            int c4  = (fi & 7) << 2;
            xf[i] = *(const float4*)(x + (m0 + row) * 512 + k0 + c4);
        }
        #pragma unroll
        for (int i = 0; i < 4; ++i) {
            int fi = i * 256 + tid;
            int nn = fi & 127;
            int kk = (fi >> 7) << 2;
            #pragma unroll
            for (int j = 0; j < 4; ++j)
                wf[i][j] = Wr[(size_t)(k0 + kk + j) * 512 + nn];
        }
    };
    auto store_tiles = [&](int buf) {
        #pragma unroll
        for (int i = 0; i < 4; ++i) {
            int fi  = i * 256 + tid;
            int row = fi >> 3;
            int c4  = (fi & 7) << 2;
            v4h h4 = { (_Float16)xf[i].x, (_Float16)xf[i].y,
                       (_Float16)xf[i].z, (_Float16)xf[i].w };
            *(v4h*)(ldsX[buf] + row * 32 + c4) = h4;
        }
        #pragma unroll
        for (int i = 0; i < 4; ++i) {
            int fi = i * 256 + tid;
            int nn = fi & 127;
            int kk = (fi >> 7) << 2;
            v4h h4 = { (_Float16)wf[i][0], (_Float16)wf[i][1],
                       (_Float16)wf[i][2], (_Float16)wf[i][3] };
            *(v4h*)(ldsW[buf] + nn * 32 + kk) = h4;
        }
    };

    load_tiles(0);
    store_tiles(0);
    __syncthreads();

    for (int kt = 0; kt < 16; ++kt) {
        const int  cur  = kt & 1;
        const bool more = kt < 15;
        if (more) load_tiles((kt + 1) * 32);  // overlap with WMMA below

        v16h bfrag[2];
        #pragma unroll
        for (int n = 0; n < 2; ++n) {
            const _Float16* p =
                ldsW[cur] + (waveN * 32 + n * 16 + (lane & 15)) * 32 + kbB;
            V16H u; u.h[0] = *(const v8h*)p; u.h[1] = *(const v8h*)(p + 16);
            bfrag[n] = u.v;
        }
        #pragma unroll
        for (int m = 0; m < 4; ++m) {
            const _Float16* p =
                ldsX[cur] + (waveM * 64 + m * 16 + (lane & 15)) * 32 + kbA;
            V16H u; u.h[0] = *(const v8h*)p; u.h[1] = *(const v8h*)(p + 16);
            v16h afrag = u.v;
            #pragma unroll
            for (int n = 0; n < 2; ++n)
                acc[m][n] = __builtin_amdgcn_wmma_f32_16x16x32_f16(
                    false, afrag, false, bfrag[n], (short)0, acc[m][n], false, false);
        }
        if (more) {
            store_tiles(1 - cur);
            __syncthreads();
        }
    }

    // ---- epilogue: gelu(val + role_bias) * W2, reduce over H ----
    float rb[2], w2[2];
    #pragma unroll
    for (int n = 0; n < 2; ++n) {
        int col = n0 + waveN * 32 + n * 16 + (lane & 15);
        rb[n] = role_bias[r * 512 + col];
        w2[n] = W2[r * 512 + col];
    }
    __syncthreads();  // tiles no longer needed; reuse barrier before ldsRed
    const int rowoff = (lane & 16) ? 8 : 0;  // C layout: lanes>=16 hold M=v+8
    #pragma unroll
    for (int m = 0; m < 4; ++m) {
        #pragma unroll
        for (int v = 0; v < 8; ++v) {
            float partial = 0.0f;
            #pragma unroll
            for (int n = 0; n < 2; ++n)
                partial += gelu_exact(acc[m][n][v] + rb[n]) * w2[n];
            #pragma unroll
            for (int off = 1; off < 16; off <<= 1)
                partial += __shfl_xor(partial, off, 32);
            if ((lane & 15) == 0)
                ldsRed[waveN][waveM * 64 + m * 16 + v + rowoff] = partial;
        }
    }
    __syncthreads();
    if (tid < 128) {
        // fixed-order sum over the 4 waveN groups -> deterministic
        float sum = ldsRed[0][tid] + ldsRed[1][tid] + ldsRed[2][tid] + ldsRed[3][tid];
        size_t mrow = m0 + tid;
        int b = (int)(mrow >> 11);
        int l = (int)(mrow & 2047);
        scores_part[(size_t)blockIdx.y * 262144 + ((size_t)b * 8 + r) * 2048 + l] = sum;
    }
}

// ---------------------------------------------------------------------------
// Kernel 3: sigmoid -> 3 binding iterations -> filler softmax -> bindings.
// One workgroup (1024 threads) per batch b. Score slab (8x2048 f32) in LDS.
// ---------------------------------------------------------------------------
__global__ void __launch_bounds__(1024)
k_bind(const float* __restrict__ scores_part,  // (4, B, R, L)
       const float* __restrict__ b2,
       const float* __restrict__ x,
       float* __restrict__ out) {  // bindings at out[0..65535]
    constexpr int R = 8, L = 2048, D = 512;
    __shared__ float s[R * L];  // 64 KB
    const int b    = blockIdx.x;
    const int tid  = threadIdx.x;
    const int lane = tid & 31;
    const float isd = 0.04419417382415922f; // 1/sqrt(512)

    for (int i = tid; i < R * L; i += 1024) {
        int r = i >> 11;
        float v = b2[r];
        #pragma unroll
        for (int p = 0; p < 4; ++p)  // fixed order -> deterministic
            v += scores_part[(size_t)p * 262144 + (size_t)b * R * L + i];
        s[i] = 1.0f / (1.0f + expf(-v));
    }
    __syncthreads();

    float rmax[R], rsum[R];
    for (int it = 0; it < 3; ++it) {
        // token-softmax row stats (scale = isd), wave-local (registers only)
        #pragma unroll
        for (int r = 0; r < R; ++r) {
            float mx = -3.0e38f;
            for (int l = lane; l < L; l += 32) mx = fmaxf(mx, s[r * L + l] * isd);
            #pragma unroll
            for (int off = 16; off; off >>= 1) mx = fmaxf(mx, __shfl_xor(mx, off, 32));
            float sm = 0.0f;
            for (int l = lane; l < L; l += 32) sm += expf(s[r * L + l] * isd - mx);
            #pragma unroll
            for (int off = 16; off; off >>= 1) sm += __shfl_xor(sm, off, 32);
            rmax[r] = mx; rsum[r] = sm;
        }
        __syncthreads();  // all stats read before columns are rewritten
        for (int l = tid; l < L; l += 1024) {
            float cv[R]; float cm = -3.0e38f;
            #pragma unroll
            for (int r = 0; r < R; ++r) { cv[r] = s[r * L + l]; cm = fmaxf(cm, cv[r]); }
            float cs = 0.0f;
            #pragma unroll
            for (int r = 0; r < R; ++r) cs += expf(cv[r] - cm);
            #pragma unroll
            for (int r = 0; r < R; ++r) {
                float rp = expf(cv[r] - cm) / cs;
                float tp = expf(cv[r] * isd - rmax[r]) / rsum[r];
                s[r * L + l] = rp * tp * cv[r];
            }
        }
        __syncthreads();
    }

    // filler softmax over l (scale = 1)
    #pragma unroll
    for (int r = 0; r < R; ++r) {
        float mx = -3.0e38f;
        for (int l = lane; l < L; l += 32) mx = fmaxf(mx, s[r * L + l]);
        #pragma unroll
        for (int off = 16; off; off >>= 1) mx = fmaxf(mx, __shfl_xor(mx, off, 32));
        float sm = 0.0f;
        for (int l = lane; l < L; l += 32) sm += expf(s[r * L + l] - mx);
        #pragma unroll
        for (int off = 16; off; off >>= 1) sm += __shfl_xor(sm, off, 32);
        rmax[r] = mx; rsum[r] = sm;
    }
    __syncthreads();
    for (int i = tid; i < R * L; i += 1024) {
        int r = i >> 11;
        s[i] = expf(s[i] - rmax[r]) / rsum[r];
    }
    __syncthreads();

    // bindings[b,r,d] = sum_l w[r,l] * x[b,l,d]
    for (int o = tid; o < R * D; o += 1024) {
        int r = o >> 9, d = o & 511;
        const float* xb = x + (size_t)b * L * D + d;
        const float* wr = s + r * L;
        float acc = 0.0f;
        #pragma unroll 4
        for (int l = 0; l < L; ++l) acc += wr[l] * xb[(size_t)l * D];
        out[((size_t)b * R + r) * D + d] = acc;
    }
}

// ---------------------------------------------------------------------------
// Kernel 4: fit head. One 512-thread workgroup per b.
// ---------------------------------------------------------------------------
__global__ void __launch_bounds__(512)
k_fit(const float* __restrict__ x,
      const float* __restrict__ schema,
      const float* __restrict__ Wf1,
      const float* __restrict__ bf1,
      const float* __restrict__ Wf2,
      const float* __restrict__ bf2,
      float* __restrict__ out) {  // fit scores at out[65536 + b]
    __shared__ float fin[512];
    __shared__ float red[16];
    const int b = blockIdx.x, t = threadIdx.x;
    const float* xb = x + (size_t)b * 2048 * 512;
    float sum = 0.0f;
    #pragma unroll 4
    for (int l = 0; l < 2048; ++l) sum += xb[(size_t)l * 512 + t];
    fin[t] = schema[t] + sum * (1.0f / 2048.0f);
    __syncthreads();
    float a = bf1[t];
    #pragma unroll 4
    for (int d = 0; d < 512; ++d) a += fin[d] * Wf1[(size_t)d * 512 + t];
    float p = gelu_exact(a) * Wf2[t];
    #pragma unroll
    for (int off = 16; off; off >>= 1) p += __shfl_xor(p, off, 32);
    if ((t & 31) == 0) red[t >> 5] = p;
    __syncthreads();
    if (t == 0) {
        float tot = 0.0f;
        #pragma unroll
        for (int i = 0; i < 16; ++i) tot += red[i];
        out[65536 + b] = 1.0f / (1.0f + expf(-(tot + bf2[0])));
    }
}

// ---------------------------------------------------------------------------
extern "C" void kernel_launch(void* const* d_in, const int* in_sizes, int n_in,
                              void* d_out, int out_size, void* d_ws, size_t ws_size,
                              hipStream_t stream) {
    const float* x        = (const float*)d_in[0];
    const float* role_emb = (const float*)d_in[1];
    const float* W1       = (const float*)d_in[2];
    const float* b1       = (const float*)d_in[3];
    const float* W2       = (const float*)d_in[4];
    const float* b2       = (const float*)d_in[5];
    const float* schema   = (const float*)d_in[6];
    const float* Wf1      = (const float*)d_in[7];
    const float* bf1      = (const float*)d_in[8];
    const float* Wf2      = (const float*)d_in[9];
    const float* bf2      = (const float*)d_in[10];
    float* out = (float*)d_out;

    float* scores_part = (float*)d_ws;            // 4 * 262144 floats
    float* role_bias   = scores_part + 4 * 262144; // 4096 floats

    k_rolebias<<<dim3(8, 2), dim3(256), 0, stream>>>(role_emb, W1, b1, role_bias);
    k_score_gemm<<<dim3(256, 4, 8), dim3(256), 0, stream>>>(x, W1, role_bias, W2, scores_part);
    k_bind<<<dim3(16), dim3(1024), 0, stream>>>(scores_part, b2, x, out);
    k_fit<<<dim3(16), dim3(512), 0, stream>>>(x, schema, Wf1, bf1, Wf2, bf2, out);
}
